// CrossAttention_8040178778262
// MI455X (gfx1250) — matrix operand order repaired and checked by reading, not compiled, not run on hardware
//
#include <hip/hip_runtime.h>

#define HEADS 8
#define DH    64
#define NN    16384
#define MM    2048
#define INNER 512
#define ATTN_SCALE 0.125f   // 1/sqrt(64)

typedef _Float16 v16h __attribute__((ext_vector_type(16)));
typedef _Float16 v8h  __attribute__((ext_vector_type(8)));
typedef float    v8f  __attribute__((ext_vector_type(8)));
typedef int      v4i  __attribute__((ext_vector_type(4)));

typedef __attribute__((address_space(1))) v4i* gv4i_ptr;   // global int4*
typedef __attribute__((address_space(3))) v4i* lv4i_ptr;   // LDS int4*

// ---------------------------------------------------------------------------
// Cross-lane reductions within 16-lane rows via DPP16 ROW_XMASK (pure VALU).
// ---------------------------------------------------------------------------
__device__ __forceinline__ float row16_max(float v) {
    int i;
    i = __builtin_amdgcn_update_dpp(0, __float_as_int(v), 0x161, 0xf, 0xf, true);
    v = fmaxf(v, __int_as_float(i));
    i = __builtin_amdgcn_update_dpp(0, __float_as_int(v), 0x162, 0xf, 0xf, true);
    v = fmaxf(v, __int_as_float(i));
    i = __builtin_amdgcn_update_dpp(0, __float_as_int(v), 0x164, 0xf, 0xf, true);
    v = fmaxf(v, __int_as_float(i));
    i = __builtin_amdgcn_update_dpp(0, __float_as_int(v), 0x168, 0xf, 0xf, true);
    v = fmaxf(v, __int_as_float(i));
    return v;
}
__device__ __forceinline__ float row16_sum(float v) {
    int i;
    i = __builtin_amdgcn_update_dpp(0, __float_as_int(v), 0x161, 0xf, 0xf, true);
    v += __int_as_float(i);
    i = __builtin_amdgcn_update_dpp(0, __float_as_int(v), 0x162, 0xf, 0xf, true);
    v += __int_as_float(i);
    i = __builtin_amdgcn_update_dpp(0, __float_as_int(v), 0x164, 0xf, 0xf, true);
    v += __int_as_float(i);
    i = __builtin_amdgcn_update_dpp(0, __float_as_int(v), 0x168, 0xf, 0xf, true);
    v += __int_as_float(i);
    return v;
}

// ---------------------------------------------------------------------------
// Async-count waits (ASYNCcnt tracks global_load_async_to_lds_*)
// ---------------------------------------------------------------------------
__device__ __forceinline__ void wait_async2() {
#if __has_builtin(__builtin_amdgcn_s_wait_asynccnt)
    __builtin_amdgcn_s_wait_asynccnt(2);
#else
    asm volatile("s_wait_asynccnt 2" ::: "memory");
#endif
}
__device__ __forceinline__ void wait_async0() {
#if __has_builtin(__builtin_amdgcn_s_wait_asynccnt)
    __builtin_amdgcn_s_wait_asynccnt(0);
#else
    asm volatile("s_wait_asynccnt 0" ::: "memory");
#endif
}

// ---------------------------------------------------------------------------
// 2x2x2 average pool of context: [128][16][32][32] -> ctxp [128][2048]
// ---------------------------------------------------------------------------
__global__ __launch_bounds__(256)
void pool_kernel(const float* __restrict__ ctx, float* __restrict__ ctxp) {
    int idx = blockIdx.x * 256 + threadIdx.x;       // 128*2048 threads
    int m = idx & (MM - 1);
    int c = idx >> 11;
    int z = m >> 8, rem = m & 255, y = rem >> 4, xw = rem & 15;
    const float* base = ctx + (size_t)c * 16384 + (size_t)(2*z) * 1024 + (2*y) * 32 + 2*xw;
    float s = 0.f;
    #pragma unroll
    for (int dz = 0; dz < 2; ++dz)
        #pragma unroll
        for (int dy = 0; dy < 2; ++dy)
            #pragma unroll
            for (int dx = 0; dx < 2; ++dx)
                s += base[dz*1024 + dy*32 + dx];
    ctxp[(size_t)c * MM + m] = s * 0.125f;
}

// ---------------------------------------------------------------------------
// Convert Wq and Wo to f16 (both 512*128 = 65536 elements)
// ---------------------------------------------------------------------------
__global__ __launch_bounds__(256)
void wconv_kernel(const float* __restrict__ Wq, const float* __restrict__ Wo,
                  _Float16* __restrict__ Wq16, _Float16* __restrict__ Wo16) {
    int idx = blockIdx.x * 256 + threadIdx.x;       // 65536 threads
    Wq16[idx] = (_Float16)Wq[idx];
    Wo16[idx] = (_Float16)Wo[idx];
}

// ---------------------------------------------------------------------------
// Q projection on WMMA: Q^T[n][o] = x^T(16n x 32c) * Wq^T(32c x 16o).
// One wave: 16 n-rows x 64 o-cols (one head's dh range), K = 128 (4 steps).
// ---------------------------------------------------------------------------
__global__ __launch_bounds__(256)
void qproj_wmma(const _Float16* __restrict__ Wq16, const float* __restrict__ x,
                _Float16* __restrict__ Q) {
    const int lane = threadIdx.x & 31;
    const int wv   = threadIdx.x >> 5;
    const int wid  = blockIdx.x * 8 + wv;           // 8192 waves
    const int og   = wid & 7;                       // head (o0 = og*64)
    const int n0   = (wid >> 3) << 4;               // 1024 n-tiles
    const int half = lane >> 4;
    const int lr   = lane & 15;

    v8f acc[4];
    #pragma unroll
    for (int t = 0; t < 4; ++t) { v8f z = {}; acc[t] = z; }

    #pragma unroll
    for (int ks = 0; ks < 4; ++ks) {
        v16h ax;
        #pragma unroll
        for (int i = 0; i < 8; ++i) {
            int c0 = ks*32 +     16*half + i;
            int c1 = ks*32 + 8 + 16*half + i;
            ax[i]     = (_Float16)x[(size_t)c0 * NN + n0 + lr];
            ax[8 + i] = (_Float16)x[(size_t)c1 * NN + n0 + lr];
        }
        #pragma unroll
        for (int t = 0; t < 4; ++t) {
            const _Float16* wrow = Wq16 + (size_t)(og*64 + t*16 + lr) * 128
                                        + ks*32 + 16*half;
            v16h bw = *(const v16h*)wrow;
            acc[t] = __builtin_amdgcn_wmma_f32_16x16x32_f16(false, ax, false, bw,
                                                            (short)0, acc[t], false, false);
        }
    }
    #pragma unroll
    for (int r = 0; r < 8; ++r) {
        int n = n0 + r + 8*half;
        _Float16* qrow = Q + ((size_t)og * NN + n) * DH;
        #pragma unroll
        for (int t = 0; t < 4; ++t)
            qrow[t*16 + lr] = (_Float16)acc[t][r];
    }
}

// ---------------------------------------------------------------------------
// K/V projection from pooled context (0.27 G MACs -> scalar is fine).
// K stored [h][m][dh], V stored [h][dh][m] for contiguous WMMA B-loads.
// ---------------------------------------------------------------------------
__global__ __launch_bounds__(256)
void kvproj_kernel(const float* __restrict__ Wk, const float* __restrict__ Wv,
                   const float* __restrict__ ctxp,
                   _Float16* __restrict__ K, _Float16* __restrict__ Vt) {
    int idx = blockIdx.x * 256 + threadIdx.x;       // 512*2048 threads
    int m = idx & (MM - 1);
    int o = idx >> 11;
    const float* wk = Wk + (size_t)o * 128;
    const float* wv = Wv + (size_t)o * 128;
    float ak = 0.f, av = 0.f;
    #pragma unroll 4
    for (int c = 0; c < 128; ++c) {
        float cv = ctxp[(size_t)c * MM + m];
        ak += wk[c] * cv;
        av += wv[c] * cv;
    }
    int h = o >> 6, d = o & 63;
    K[((size_t)h * MM + m) * DH + d] = (_Float16)ak;
    Vt[(size_t)o * MM + m] = (_Float16)av;          // o == h*64+d -> [h][dh][m]
}

// ---------------------------------------------------------------------------
// Flash attention core. Block = 8 waves, all on the SAME head, 8 adjacent
// query tiles. K/V tiles are staged into LDS once per block via async
// global->LDS DMA, double-buffered (issue tile i+1, wait ASYNCcnt<=2 for
// tile i, barrier, compute). B-operands feed from LDS; softmax stats via
// DPP ROW_XMASK; P staged through per-wave LDS slab.
// ---------------------------------------------------------------------------
__global__ __launch_bounds__(256)
void attn_kernel(const _Float16* __restrict__ Q, const _Float16* __restrict__ K,
                 const _Float16* __restrict__ Vt, _Float16* __restrict__ O) {
    __shared__ _Float16 Kt[2][32][64];              // K tile:  32 m x 64 d (4KB x2)
    __shared__ _Float16 Vl[2][64][32];              // V tile:  64 d x 32 m (4KB x2)
    __shared__ _Float16 Plds[8][16][32];            // per-wave P staging (8KB)

    const int tid  = threadIdx.x;
    const int lane = tid & 31;
    const int wv   = tid >> 5;
    const int head = blockIdx.x >> 7;               // 128 blocks per head
    const int n0   = ((((blockIdx.x & 127) << 3) | wv)) << 4;
    const int half = lane >> 4;
    const int lr   = lane & 15;

    const _Float16* kbase = K  + (size_t)head * MM * DH;
    const _Float16* vbase = Vt + (size_t)head * DH * MM;

    // cooperative 16B-per-thread tile copy: K 32x64 (8 chunks/row), V 64x32 (4)
    const int krow_c = tid >> 3, kchk = (tid & 7) * 8;
    const int vrow_c = tid >> 2, vchk = (tid & 3) * 8;

    auto loadTile = [&](int buf, int m0) {
        const _Float16* ksrc = kbase + (size_t)(m0 + krow_c) * DH + kchk;
        const _Float16* vsrc = vbase + (size_t)vrow_c * MM + m0 + vchk;
        _Float16* kdst = &Kt[buf][krow_c][kchk];
        _Float16* vdst = &Vl[buf][vrow_c][vchk];
#if __has_builtin(__builtin_amdgcn_global_load_async_to_lds_b128)
        __builtin_amdgcn_global_load_async_to_lds_b128(
            (gv4i_ptr)ksrc, (lv4i_ptr)kdst, 0, 0);
        __builtin_amdgcn_global_load_async_to_lds_b128(
            (gv4i_ptr)vsrc, (lv4i_ptr)vdst, 0, 0);
#else
        *(v8h*)kdst = *(const v8h*)ksrc;
        *(v8h*)vdst = *(const v8h*)vsrc;
#endif
    };

    loadTile(0, 0);                                  // prologue: tile 0 in flight

    // --- A-operands for Q tile (16 rows x 64 k), contiguous k order (k = 16*half + s per 32-k chunk) ---
    const _Float16* qrow = Q + ((size_t)head * NN + (n0 + lr)) * DH;
    v16h aq0, aq1;
    {
        v8h c0 = *(const v8h*)(qrow +      16*half);
        v8h c1 = *(const v8h*)(qrow +  8 + 16*half);
        v8h c2 = *(const v8h*)(qrow + 32 + 16*half);
        v8h c3 = *(const v8h*)(qrow + 40 + 16*half);
        #pragma unroll
        for (int i = 0; i < 8; ++i) {
            aq0[i] = c0[i]; aq0[8+i] = c1[i];
            aq1[i] = c2[i]; aq1[8+i] = c3[i];
        }
    }

    v8f acc[4];
    #pragma unroll
    for (int t = 0; t < 4; ++t) { v8f z = {}; acc[t] = z; }
    float rowmax[8], rowsum[8];
    #pragma unroll
    for (int r = 0; r < 8; ++r) { rowmax[r] = -1e30f; rowsum[r] = 0.f; }

    for (int it = 0; it < MM / 32; ++it) {
        const int buf = it & 1;
        if (it + 1 < MM / 32) {
            loadTile(buf ^ 1, (it + 1) * 32);        // issue next tile's DMA
            wait_async2();                           // oldest 2 (tile it) done
        } else {
            wait_async0();
        }
        __syncthreads();                             // tile visible to all waves

        // ---- scores: S[16n x 32m] = Q(16x64) * K^T, two 16x16 f32 tiles ----
        v8f s[2];
        #pragma unroll
        for (int mh = 0; mh < 2; ++mh) {
            const _Float16* krow = &Kt[buf][16*mh + lr][0];
            v16h b0 = *(const v16h*)(krow +      16*half);   // k = 0..31
            v16h b1 = *(const v16h*)(krow + 32 + 16*half);   // k = 32..63
            v8f z = {};
            z = __builtin_amdgcn_wmma_f32_16x16x32_f16(false, aq0, false, b0,
                                                       (short)0, z, false, false);
            z = __builtin_amdgcn_wmma_f32_16x16x32_f16(false, aq1, false, b1,
                                                       (short)0, z, false, false);
            s[mh] = z;
        }

        // ---- online softmax (row = vgpr idx + 8*half; cols = 16 lanes) ----
        _Float16 p0h[8], p1h[8];
        #pragma unroll
        for (int r = 0; r < 8; ++r) {
            float s0r = s[0][r] * ATTN_SCALE;
            float s1r = s[1][r] * ATTN_SCALE;
            float tmax = row16_max(fmaxf(s0r, s1r));
            float mnew = fmaxf(rowmax[r], tmax);
            float corr = __expf(rowmax[r] - mnew);
            float p0 = __expf(s0r - mnew);
            float p1 = __expf(s1r - mnew);
            float ps = row16_sum(p0 + p1);
            rowsum[r] = rowsum[r] * corr + ps;
            rowmax[r] = mnew;
            #pragma unroll
            for (int t = 0; t < 4; ++t) acc[t][r] *= corr;
            p0h[r] = (_Float16)p0;
            p1h[r] = (_Float16)p1;
        }

        // ---- stage P (D layout -> LDS row-major -> A layout), same-wave ----
        #pragma unroll
        for (int r = 0; r < 8; ++r) {
            Plds[wv][r + 8*half][lr]      = p0h[r];
            Plds[wv][r + 8*half][16 + lr] = p1h[r];
        }
        asm volatile("s_wait_dscnt 0" ::: "memory");
        v16h ap;
        {
            v8h c0 = *(const v8h*)&Plds[wv][lr][16*half];
            v8h c1 = *(const v8h*)&Plds[wv][lr][8 + 16*half];
            #pragma unroll
            for (int i = 0; i < 8; ++i) { ap[i] = c0[i]; ap[8+i] = c1[i]; }
        }

        // ---- out += P(16x32m) * V(32m x 64d): four 16x16 d tiles ----
        #pragma unroll
        for (int t = 0; t < 4; ++t) {
            v16h bv = *(const v16h*)&Vl[buf][t*16 + lr][16*half];
            acc[t] = __builtin_amdgcn_wmma_f32_16x16x32_f16(false, ap, false, bv,
                                                            (short)0, acc[t], false, false);
        }
        __syncthreads();                             // all waves done with buf
    }

    // ---- normalize and store O[n][512] (contiguous 32B chunks per row) ----
    #pragma unroll
    for (int r = 0; r < 8; ++r) {
        float inv = 1.0f / rowsum[r];
        int n = n0 + r + 8*half;
        _Float16* orow = O + (size_t)n * INNER + head * DH;
        #pragma unroll
        for (int t = 0; t < 4; ++t)
            orow[t*16 + lr] = (_Float16)(acc[t][r] * inv);
    }
}

// ---------------------------------------------------------------------------
// Output projection on WMMA: y^T[n][o] = O^T(16n x 32c) * Wo^T(32c x 16o).
// One wave: 16 n-rows x 64 o-cols, K = 512 (16 steps). Contiguous operands.
// ---------------------------------------------------------------------------
__global__ __launch_bounds__(256)
void oproj_wmma(const _Float16* __restrict__ Wo16, const _Float16* __restrict__ O,
                float* __restrict__ y) {
    const int lane  = threadIdx.x & 31;
    const int wv    = threadIdx.x >> 5;
    const int wid   = blockIdx.x * 8 + wv;          // 2048 waves
    const int ohalf = wid & 1;                      // o0 = ohalf*64
    const int n0    = (wid >> 1) << 4;              // 1024 n-tiles
    const int half  = lane >> 4;
    const int lr    = lane & 15;

    v8f acc[4];
    #pragma unroll
    for (int t = 0; t < 4; ++t) { v8f z = {}; acc[t] = z; }

    for (int ks = 0; ks < 16; ++ks) {
        const _Float16* orow = O + (size_t)(n0 + lr) * INNER + ks*32;
        v16h ao;
        {
            v8h c0 = *(const v8h*)(orow +     16*half);
            v8h c1 = *(const v8h*)(orow + 8 + 16*half);
            #pragma unroll
            for (int i = 0; i < 8; ++i) { ao[i] = c0[i]; ao[8+i] = c1[i]; }
        }
        #pragma unroll
        for (int t = 0; t < 4; ++t) {
            const _Float16* wrow = Wo16 + (size_t)(ohalf*64 + t*16 + lr) * INNER
                                        + ks*32 + 16*half;
            v16h bw = *(const v16h*)wrow;
            acc[t] = __builtin_amdgcn_wmma_f32_16x16x32_f16(false, ao, false, bw,
                                                            (short)0, acc[t], false, false);
        }
    }
    #pragma unroll
    for (int r = 0; r < 8; ++r) {
        int n = n0 + r + 8*half;
        #pragma unroll
        for (int t = 0; t < 4; ++t)
            y[(size_t)(ohalf*64 + t*16 + lr) * NN + n] = acc[t][r];
    }
}

// ---------------------------------------------------------------------------
extern "C" void kernel_launch(void* const* d_in, const int* in_sizes, int n_in,
                              void* d_out, int out_size, void* d_ws, size_t ws_size,
                              hipStream_t stream) {
    const float* x   = (const float*)d_in[0];
    const float* ctx = (const float*)d_in[1];
    const float* Wq  = (const float*)d_in[2];
    const float* Wk  = (const float*)d_in[3];
    const float* Wv  = (const float*)d_in[4];
    const float* Wo  = (const float*)d_in[5];
    float* y = (float*)d_out;

    char* ws = (char*)d_ws;
    float*    ctxp = (float*)ws;     ws += (size_t)128 * MM * 4;            // 1 MB
    _Float16* Q    = (_Float16*)ws;  ws += (size_t)HEADS * NN * DH * 2;     // 16 MB
    _Float16* K    = (_Float16*)ws;  ws += (size_t)HEADS * MM * DH * 2;     // 2 MB
    _Float16* Vt   = (_Float16*)ws;  ws += (size_t)HEADS * MM * DH * 2;     // 2 MB
    _Float16* O    = (_Float16*)ws;  ws += (size_t)NN * INNER * 2;          // 16 MB
    _Float16* Wq16 = (_Float16*)ws;  ws += (size_t)INNER * 128 * 2;         // 128 KB
    _Float16* Wo16 = (_Float16*)ws;                                         // 128 KB

    pool_kernel  <<<(128 * MM) / 256,          256, 0, stream>>>(ctx, ctxp);
    wconv_kernel <<<(INNER * 128) / 256,       256, 0, stream>>>(Wq, Wo, Wq16, Wo16);
    qproj_wmma   <<<(HEADS * (NN/16)) / 8,     256, 0, stream>>>(Wq16, x, Q);
    kvproj_kernel<<<(INNER * MM) / 256,        256, 0, stream>>>(Wk, Wv, ctxp, K, Vt);
    attn_kernel  <<<(HEADS * (NN/16)) / 8,     256, 0, stream>>>(Q, K, Vt, O);
    oproj_wmma   <<<(2 * (NN/16)) / 8,         256, 0, stream>>>(Wo16, O, y);
}
